// GCN_9801115370057
// MI455X (gfx1250) — compile-verified, matrix operand order checked by reading, not run on hardware
//
#include <hip/hip_runtime.h>
#include <hip/hip_bf16.h>

#define N_NODES 50000
#define N_EDGES 800000
#define IN_FT   512
#define H_FT    256
#define OUT_FT  40
#define M_TILES (N_NODES / 16)   // 3125 exactly

typedef __bf16 bf16_t;
typedef bf16_t v16bf __attribute__((ext_vector_type(16)));
typedef float  v8f   __attribute__((ext_vector_type(8)));

union Frag {
    v16bf v;
    unsigned short s[16];
    uint4 q[2];          // 2 x 16B for b128 loads
};

// float -> bf16 round-to-nearest-even (bit trick)
__device__ __forceinline__ unsigned short f2bf(float f) {
    unsigned u = __float_as_uint(f);
    u += 0x7fffu + ((u >> 16) & 1u);
    return (unsigned short)(u >> 16);
}
__device__ __forceinline__ float bf2f(unsigned short h) {
    return __uint_as_float((unsigned)h << 16);
}
__device__ __forceinline__ void split(float v, unsigned short& hi, unsigned short& lo) {
    hi = f2bf(v);
    lo = f2bf(v - bf2f(hi));
}

// ---------------------------------------------------------------------------
// split_pack: fp32[total8*8] -> hi/lo bf16 planes (row-major), optional relu.
// 8 elements per thread, b128 in / b128 out.
// ---------------------------------------------------------------------------
__global__ __launch_bounds__(256) void split_pack_kernel(
    const float* __restrict__ in, unsigned short* __restrict__ hi,
    unsigned short* __restrict__ lo, int total8, int do_relu) {
    int idx = blockIdx.x * 256 + threadIdx.x;
    if (idx >= total8) return;
    const float4* p = reinterpret_cast<const float4*>(in) + (size_t)idx * 2;
    float4 a = p[0], b = p[1];
    float v[8] = {a.x, a.y, a.z, a.w, b.x, b.y, b.z, b.w};
    union { uint4 q; unsigned short s[8]; } H, L;
#pragma unroll
    for (int j = 0; j < 8; ++j) {
        float f = do_relu ? fmaxf(v[j], 0.f) : v[j];
        split(f, H.s[j], L.s[j]);
    }
    reinterpret_cast<uint4*>(hi)[idx] = H.q;
    reinterpret_cast<uint4*>(lo)[idx] = L.q;
}

// ---------------------------------------------------------------------------
// pack_b: repack W[K][Ncols] (row-major fp32) into WMMA B-operand order:
//   dst[((ks*nTiles + nt)*32 + lane)*16 + j] = split(W[k][col]),
//   k = ks*32 + klo(lane) + j + (j&8), col = nt*16 + (lane&15)  (0 if >= Ncols)
// One thread per (ks, nt, lane); B fragment load becomes 2 coalesced b128s.
// ---------------------------------------------------------------------------
__global__ __launch_bounds__(256) void pack_b_kernel(
    const float* __restrict__ W, unsigned short* __restrict__ hi,
    unsigned short* __restrict__ lo, int Kdim, int Ncols, int nTiles) {
    int nK = Kdim / 32;
    int total = nK * nTiles * 32;
    int idx = blockIdx.x * 256 + threadIdx.x;
    if (idx >= total) return;
    int lane = idx & 31;
    int nt = (idx >> 5) % nTiles;
    int ks = (idx >> 5) / nTiles;
    int col = nt * 16 + (lane & 15);
    int klo = (lane >> 4) << 3;
    Frag H, L;
#pragma unroll
    for (int j = 0; j < 16; ++j) {
        int k = ks * 32 + klo + j + (j & 8);
        float v = (col < Ncols) ? W[(size_t)k * Ncols + col] : 0.f;
        split(v, H.s[j], L.s[j]);
    }
    uint4* ho = reinterpret_cast<uint4*>(hi) + (size_t)idx * 2;
    uint4* lop = reinterpret_cast<uint4*>(lo) + (size_t)idx * 2;
    ho[0] = H.q[0]; ho[1] = H.q[1];
    lop[0] = L.q[0]; lop[1] = L.q[1];
}

// ---------------------------------------------------------------------------
// GEMM1: support1[50000,256] = x @ W1 + b1   (packed bf16 hi/lo operands)
// block = 128 thr (4 waves); wave w -> N-tiles [w*4..w*4+3]; block -> 4 M-tiles
// ---------------------------------------------------------------------------
__global__ __launch_bounds__(128) void gemm1_kernel(
    const unsigned short* __restrict__ Xhi, const unsigned short* __restrict__ Xlo,
    const unsigned short* __restrict__ Bhi, const unsigned short* __restrict__ Blo,
    const float* __restrict__ bias, float* __restrict__ Out) {
    const int lane  = threadIdx.x & 31;
    const int wave  = threadIdx.x >> 5;
    const int lm    = lane & 15;
    const int klo   = (lane >> 4) << 3;
    const int mrow8 = (lane >> 4) << 3;
    const int mbase = blockIdx.x * (4 * 16);
    const int nt0   = wave * 4;                 // global N-tile base (of 16)

    v8f acc[4][4];
#pragma unroll
    for (int nt = 0; nt < 4; ++nt) {
        float bv = bias[(nt0 + nt) * 16 + lm];
#pragma unroll
        for (int mt = 0; mt < 4; ++mt)
#pragma unroll
            for (int i = 0; i < 8; ++i) acc[mt][nt][i] = bv;
    }

    for (int ksi = 0; ksi < IN_FT / 32; ++ksi) {
        Frag ahi[4], alo[4];
#pragma unroll
        for (int mt = 0; mt < 4; ++mt) {
            int row = mbase + mt * 16 + lm;
            if (row >= N_NODES) row = N_NODES - 1;   // clamp; EXEC stays all-1s
            size_t e = (size_t)row * IN_FT + ksi * 32 + klo;
            const uint4* ph = reinterpret_cast<const uint4*>(Xhi + e);
            const uint4* pl = reinterpret_cast<const uint4*>(Xlo + e);
            ahi[mt].q[0] = ph[0]; ahi[mt].q[1] = ph[2];   // K: klo..+7, klo+16..+23
            alo[mt].q[0] = pl[0]; alo[mt].q[1] = pl[2];
        }
#pragma unroll
        for (int nt = 0; nt < 4; ++nt) {
            size_t bi = (((size_t)ksi * 16 + (nt0 + nt)) * 32 + lane) * 2; // uint4 units
            Frag bh, bl;
            bh.q[0] = reinterpret_cast<const uint4*>(Bhi)[bi];
            bh.q[1] = reinterpret_cast<const uint4*>(Bhi)[bi + 1];
            bl.q[0] = reinterpret_cast<const uint4*>(Blo)[bi];
            bl.q[1] = reinterpret_cast<const uint4*>(Blo)[bi + 1];
#pragma unroll
            for (int mt = 0; mt < 4; ++mt) {
                acc[mt][nt] = __builtin_amdgcn_wmma_f32_16x16x32_bf16(
                    false, ahi[mt].v, false, bh.v, (short)0, acc[mt][nt], false, false);
                acc[mt][nt] = __builtin_amdgcn_wmma_f32_16x16x32_bf16(
                    false, ahi[mt].v, false, bl.v, (short)0, acc[mt][nt], false, false);
                acc[mt][nt] = __builtin_amdgcn_wmma_f32_16x16x32_bf16(
                    false, alo[mt].v, false, bh.v, (short)0, acc[mt][nt], false, false);
            }
        }
    }

#pragma unroll
    for (int mt = 0; mt < 4; ++mt)
#pragma unroll
        for (int nt = 0; nt < 4; ++nt) {
            int col = (nt0 + nt) * 16 + lm;
#pragma unroll
            for (int r = 0; r < 8; ++r) {
                int row = mbase + mt * 16 + mrow8 + r;
                if (row < N_NODES) Out[(size_t)row * H_FT + col] = acc[mt][nt][r];
            }
        }
}

// ---------------------------------------------------------------------------
// SpMM1: h1[dst] += w * support1[src]  (F=256, 64 lanes/edge, float4 + atomics)
// ---------------------------------------------------------------------------
__global__ __launch_bounds__(256) void spmm1_kernel(
    const int* __restrict__ src, const int* __restrict__ dst,
    const float* __restrict__ ew, const float* __restrict__ Hin,
    float* __restrict__ Hout) {
    int t = threadIdx.x;
    int e = blockIdx.x * 4 + (t >> 6);
    if (e >= N_EDGES) return;
    int l = t & 63;
    int s = src[e], d = dst[e];
    float w = ew[e];
    float4 v = reinterpret_cast<const float4*>(Hin + (size_t)s * H_FT)[l];
    float* o = Hout + (size_t)d * H_FT + l * 4;
    atomicAdd(o + 0, v.x * w);
    atomicAdd(o + 1, v.y * w);
    atomicAdd(o + 2, v.z * w);
    atomicAdd(o + 3, v.w * w);
}

// ---------------------------------------------------------------------------
// GEMM2: support2[50000,40] = relu(h1) @ W2 + b2  (A pre-packed with relu)
// block = 256 thr (8 waves); wave -> 1 M-tile x 3 N-tiles (cols padded to 48)
// ---------------------------------------------------------------------------
__global__ __launch_bounds__(256) void gemm2_kernel(
    const unsigned short* __restrict__ Ahi, const unsigned short* __restrict__ Alo,
    const unsigned short* __restrict__ Bhi, const unsigned short* __restrict__ Blo,
    const float* __restrict__ bias, float* __restrict__ Out) {
    const int lane  = threadIdx.x & 31;
    const int wave  = threadIdx.x >> 5;
    const int lm    = lane & 15;
    const int klo   = (lane >> 4) << 3;
    const int mrow8 = (lane >> 4) << 3;
    const int mtile = blockIdx.x * 8 + wave;
    if (mtile >= M_TILES) return;      // uniform per wave

    v8f acc[3];
#pragma unroll
    for (int nt = 0; nt < 3; ++nt) {
        int col = nt * 16 + lm;
        float bv = (col < OUT_FT) ? bias[col] : 0.f;
#pragma unroll
        for (int i = 0; i < 8; ++i) acc[nt][i] = bv;
    }

    const int row = mtile * 16 + lm;   // < 50000 guaranteed
    for (int ksi = 0; ksi < H_FT / 32; ++ksi) {
        size_t e = (size_t)row * H_FT + ksi * 32 + klo;
        Frag ah, al;
        const uint4* ph = reinterpret_cast<const uint4*>(Ahi + e);
        const uint4* pl = reinterpret_cast<const uint4*>(Alo + e);
        ah.q[0] = ph[0]; ah.q[1] = ph[2];
        al.q[0] = pl[0]; al.q[1] = pl[2];
#pragma unroll
        for (int nt = 0; nt < 3; ++nt) {
            size_t bi = (((size_t)ksi * 3 + nt) * 32 + lane) * 2;
            Frag bh, bl;
            bh.q[0] = reinterpret_cast<const uint4*>(Bhi)[bi];
            bh.q[1] = reinterpret_cast<const uint4*>(Bhi)[bi + 1];
            bl.q[0] = reinterpret_cast<const uint4*>(Blo)[bi];
            bl.q[1] = reinterpret_cast<const uint4*>(Blo)[bi + 1];
            acc[nt] = __builtin_amdgcn_wmma_f32_16x16x32_bf16(
                false, ah.v, false, bh.v, (short)0, acc[nt], false, false);
            acc[nt] = __builtin_amdgcn_wmma_f32_16x16x32_bf16(
                false, ah.v, false, bl.v, (short)0, acc[nt], false, false);
            acc[nt] = __builtin_amdgcn_wmma_f32_16x16x32_bf16(
                false, al.v, false, bh.v, (short)0, acc[nt], false, false);
        }
    }

#pragma unroll
    for (int nt = 0; nt < 3; ++nt) {
        int col = nt * 16 + lm;
        if (col < OUT_FT) {
#pragma unroll
            for (int r = 0; r < 8; ++r) {
                int orow = mtile * 16 + mrow8 + r;
                Out[(size_t)orow * OUT_FT + col] = acc[nt][r];
            }
        }
    }
}

// ---------------------------------------------------------------------------
// SpMM2: out[dst] += w * support2[src]  (F=40, one thread per (edge,feature))
// ---------------------------------------------------------------------------
__global__ __launch_bounds__(256) void spmm2_kernel(
    const int* __restrict__ src, const int* __restrict__ dst,
    const float* __restrict__ ew, const float* __restrict__ Hin,
    float* __restrict__ Out) {
    int idx = blockIdx.x * 256 + threadIdx.x;
    if (idx >= N_EDGES * OUT_FT) return;
    int e = idx / OUT_FT;
    int f = idx - e * OUT_FT;
    int s = src[e], d = dst[e];
    atomicAdd(&Out[(size_t)d * OUT_FT + f], ew[e] * Hin[(size_t)s * OUT_FT + f]);
}

// ---------------------------------------------------------------------------
// relu + log_softmax in place over rows of 40; one wave32 per row
// ---------------------------------------------------------------------------
__global__ __launch_bounds__(256) void lsm_kernel(float* __restrict__ Out) {
    int lane = threadIdx.x & 31;
    int row  = blockIdx.x * 8 + (threadIdx.x >> 5);
    if (row >= N_NODES) return;    // uniform per wave
    float* p = Out + (size_t)row * OUT_FT;
    float a = fmaxf(p[lane], 0.f);
    bool has2 = (lane + 32) < OUT_FT;
    float b = has2 ? fmaxf(p[lane + 32], 0.f) : 0.f;

    float m = a;
    if (has2) m = fmaxf(m, b);
#pragma unroll
    for (int off = 16; off >= 1; off >>= 1)
        m = fmaxf(m, __shfl_xor(m, off, 32));

    float s = expf(a - m) + (has2 ? expf(b - m) : 0.f);
#pragma unroll
    for (int off = 16; off >= 1; off >>= 1)
        s += __shfl_xor(s, off, 32);

    float L = m + logf(s);
    p[lane] = a - L;
    if (has2) p[lane + 32] = b - L;
}

// ---------------------------------------------------------------------------
extern "C" void kernel_launch(void* const* d_in, const int* in_sizes, int n_in,
                              void* d_out, int out_size, void* d_ws, size_t ws_size,
                              hipStream_t stream) {
    const float* x    = (const float*)d_in[0];
    const int*   esrc = (const int*)d_in[1];
    const int*   edst = (const int*)d_in[2];
    const float* ew   = (const float*)d_in[3];
    const float* W1   = (const float*)d_in[4];
    const float* b1   = (const float*)d_in[5];
    const float* W2   = (const float*)d_in[6];
    const float* b2   = (const float*)d_in[7];
    float* out = (float*)d_out;

    char* ws = (char*)d_ws;
    const size_t SZ_XPLANE = (size_t)N_NODES * IN_FT * 2;   // 51.2 MB (bf16 plane of x)
    const size_t SZ_HPLANE = (size_t)N_NODES * H_FT * 2;    // 25.6 MB (bf16 plane of h1)

    // Phase 1 views
    unsigned short* Xhi = (unsigned short*)(ws);
    unsigned short* Xlo = (unsigned short*)(ws + SZ_XPLANE);
    float* support1     = (float*)(ws + 2 * SZ_XPLANE);     // 51.2 MB
    // Phase 2 views (reuse Xhi/Xlo space once GEMM1 is done)
    float* h1           = (float*)(ws);                     // 51.2 MB (over Xhi)
    unsigned short* H1hi = (unsigned short*)(ws + SZ_XPLANE);               // over Xlo
    unsigned short* H1lo = (unsigned short*)(ws + SZ_XPLANE + SZ_HPLANE);
    float* support2     = (float*)(ws + 2 * SZ_XPLANE);     // over support1 (dead)
    // Weight packs (persistent, beyond support1 region)
    char* wp = ws + 2 * SZ_XPLANE + (size_t)N_NODES * H_FT * 4;
    unsigned short* W1hiP = (unsigned short*)wp;                  // 16*16*32*16 el
    unsigned short* W1loP = W1hiP + (size_t)16 * 16 * 32 * 16;
    unsigned short* W2hiP = W1loP + (size_t)16 * 16 * 32 * 16;    // 8*3*32*16 el
    unsigned short* W2loP = W2hiP + (size_t)8 * 3 * 32 * 16;

    // --- pack operands ---
    {
        int total8 = N_NODES * IN_FT / 8;    // 3.2M
        split_pack_kernel<<<(total8 + 255) / 256, 256, 0, stream>>>(
            x, Xhi, Xlo, total8, 0);
        pack_b_kernel<<<(16 * 16 * 32 + 255) / 256, 256, 0, stream>>>(
            W1, W1hiP, W1loP, IN_FT, H_FT, 16);
        pack_b_kernel<<<(8 * 3 * 32 + 255) / 256, 256, 0, stream>>>(
            W2, W2hiP, W2loP, H_FT, OUT_FT, 3);
    }

    // --- layer 1 ---
    gemm1_kernel<<<(M_TILES + 3) / 4, 128, 0, stream>>>(
        Xhi, Xlo, W1hiP, W1loP, b1, support1);
    hipMemsetAsync(h1, 0, (size_t)N_NODES * H_FT * sizeof(float), stream);
    spmm1_kernel<<<N_EDGES / 4, 256, 0, stream>>>(esrc, edst, ew, support1, h1);

    // --- layer 2 (relu fused into the split-pack of h1) ---
    {
        int total8 = N_NODES * H_FT / 8;     // 1.6M
        split_pack_kernel<<<(total8 + 255) / 256, 256, 0, stream>>>(
            h1, H1hi, H1lo, total8, 1);
    }
    gemm2_kernel<<<(M_TILES + 7) / 8, 256, 0, stream>>>(
        H1hi, H1lo, W2hiP, W2loP, b2, support2);
    hipMemsetAsync(out, 0, (size_t)N_NODES * OUT_FT * sizeof(float), stream);
    spmm2_kernel<<<(N_EDGES * OUT_FT + 255) / 256, 256, 0, stream>>>(
        esrc, edst, ew, support2, out);

    // --- relu + log_softmax in place ---
    lsm_kernel<<<(N_NODES + 7) / 8, 256, 0, stream>>>(out);
}
